// STPEBlock_68204080660811
// MI455X (gfx1250) — matrix operand order repaired and checked by reading, not compiled, not run on hardware
//
#include <hip/hip_runtime.h>
#include <hip/hip_bf16.h>
#include <math.h>

typedef _Float16 half_t;
typedef __attribute__((ext_vector_type(16))) _Float16 v16h;
typedef __attribute__((ext_vector_type(8)))  _Float16 v8h;
typedef __attribute__((ext_vector_type(8)))  float    v8f;

#define DMODEL 128
#define BATCH  16
#define KF     24
#define NTOK   24   // 3*KC coarse tokens == KF fine tokens == 24

// ---------------------------------------------------------------------------
// proj_in:  t[b,s,d] = sum_c x[b,c,s] * win[d,c]
// WMMA: A = X^T tile (M=s,16 x K=c,32), B = win^T tile (K=c,32 x N=d,16)
// Full win (f16) resident in LDS. C,S compile-time so all A-loads use
// immediate offsets (16 loads in flight, single wait).
// ---------------------------------------------------------------------------
template <int C, int S>
__global__ __launch_bounds__(256) void proj_in_wmma(
    const float* __restrict__ x, const float* __restrict__ win,
    float* __restrict__ t)
{
  extern __shared__ float4 dynsmem[];
  half_t* lw = (half_t*)dynsmem;
  const int tid = threadIdx.x;
  for (int i = tid; i < DMODEL * C; i += blockDim.x) lw[i] = (half_t)win[i];
  __syncthreads();

  const int lane = tid & 31;
  const int wave = tid >> 5;
  const int r    = lane & 15;
  const int hi   = lane >> 4;
  constexpr int tilesPerB = S >> 4;
  constexpr int nTiles = BATCH * tilesPerB;

  for (int tile = blockIdx.x * 8 + wave; tile < nTiles; tile += gridDim.x * 8) {
    const int b  = tile / tilesPerB;
    const int s0 = (tile % tilesPerB) << 4;
    const int srow = s0 + r;
    // fold hi*8*S into base -> per-e offsets are compile-time constants
    const float* pa = x + (size_t)b * C * S + (size_t)hi * 8 * S + srow;

    v8f acc[8];
    #pragma unroll
    for (int n = 0; n < 8; ++n)
      #pragma unroll
      for (int j = 0; j < 8; ++j) acc[n][j] = 0.0f;

    for (int c0 = 0; c0 < C; c0 += 32, pa += (size_t)32 * S) {
      v16h a;
      #pragma unroll
      for (int e = 0; e < 8; ++e) {
        a[e]     = (half_t)pa[(size_t)e * S];          // K = c0 + hi*8 + e
        a[e + 8] = (half_t)pa[(size_t)(16 + e) * S];   // K = c0 + 16 + hi*8 + e
      }
      #pragma unroll
      for (int n = 0; n < 8; ++n) {
        const int off = (n * 16 + r) * C + c0 + hi * 8;  // K = hi*8+j and 16+hi*8+j
        v8h b0 = *(const v8h*)(lw + off);
        v8h b1 = *(const v8h*)(lw + off + 16);
        v16h bf;
        #pragma unroll
        for (int j = 0; j < 8; ++j) { bf[j] = b0[j]; bf[j + 8] = b1[j]; }
        acc[n] = __builtin_amdgcn_wmma_f32_16x16x32_f16(
            false, a, false, bf, (short)0, acc[n], false, false);
      }
    }
    // D layout: M = rr + hi*8 (s), N = r (d within 16-tile)
    float* tb = t + ((size_t)b * S + s0 + hi * 8) * DMODEL;
    #pragma unroll
    for (int rr = 0; rr < 8; ++rr)
      #pragma unroll
      for (int n = 0; n < 8; ++n)
        tb[(size_t)rr * DMODEL + n * 16 + r] = acc[n][rr];
  }
}

// ---------------------------------------------------------------------------
// proj_out: out[b,c,s] = sum_d t[b,s,d] * wout[c,d]
// A = t tile (M=s x K=d), preloaded once per s-tile; B = wout^T from LDS.
// ---------------------------------------------------------------------------
template <int C, int S>
__global__ __launch_bounds__(256) void proj_out_wmma(
    const float* __restrict__ t, const float* __restrict__ wout,
    float* __restrict__ out)
{
  extern __shared__ float4 dynsmem2[];
  half_t* lw = (half_t*)dynsmem2;
  const int tid = threadIdx.x;
  for (int i = tid; i < C * DMODEL; i += blockDim.x) lw[i] = (half_t)wout[i];
  __syncthreads();

  const int lane = tid & 31;
  const int wave = tid >> 5;
  const int r    = lane & 15;
  const int hi   = lane >> 4;
  constexpr int tilesPerB = S >> 4;
  constexpr int nTiles = BATCH * tilesPerB;

  for (int tile = blockIdx.x * 8 + wave; tile < nTiles; tile += gridDim.x * 8) {
    const int b  = tile / tilesPerB;
    const int s0 = (tile % tilesPerB) << 4;
    const int srow = s0 + r;
    const float* tr = t + ((size_t)b * S + srow) * DMODEL;

    v16h a[4];                     // 4 K-slices of 32 over d = 0..127
    #pragma unroll
    for (int kk = 0; kk < 4; ++kk) {
      const int dbase = kk * 32 + hi * 8;
      const float4 x0 = *(const float4*)(tr + dbase);
      const float4 x1 = *(const float4*)(tr + dbase + 4);
      const float4 x2 = *(const float4*)(tr + dbase + 16);
      const float4 x3 = *(const float4*)(tr + dbase + 20);
      a[kk][0]=(half_t)x0.x;  a[kk][1]=(half_t)x0.y;  a[kk][2]=(half_t)x0.z;  a[kk][3]=(half_t)x0.w;
      a[kk][4]=(half_t)x1.x;  a[kk][5]=(half_t)x1.y;  a[kk][6]=(half_t)x1.z;  a[kk][7]=(half_t)x1.w;
      a[kk][8]=(half_t)x2.x;  a[kk][9]=(half_t)x2.y;  a[kk][10]=(half_t)x2.z; a[kk][11]=(half_t)x2.w;
      a[kk][12]=(half_t)x3.x; a[kk][13]=(half_t)x3.y; a[kk][14]=(half_t)x3.z; a[kk][15]=(half_t)x3.w;
    }

    constexpr int nCt = C >> 4;
    for (int ct = 0; ct < nCt; ++ct) {
      const int c = ct * 16 + r;            // N index for this lane
      v8f acc;
      #pragma unroll
      for (int j = 0; j < 8; ++j) acc[j] = 0.0f;
      #pragma unroll
      for (int kk = 0; kk < 4; ++kk) {
        const int off = c * DMODEL + kk * 32 + hi * 8;
        v8h b0 = *(const v8h*)(lw + off);
        v8h b1 = *(const v8h*)(lw + off + 16);
        v16h bf;
        #pragma unroll
        for (int j = 0; j < 8; ++j) { bf[j] = b0[j]; bf[j + 8] = b1[j]; }
        acc = __builtin_amdgcn_wmma_f32_16x16x32_f16(
            false, a[kk], false, bf, (short)0, acc, false, false);
      }
      // lane holds 8 consecutive s at fixed c -> two b128 stores
      float* op = out + ((size_t)b * C + c) * S + s0 + hi * 8;
      *(float4*)op       = make_float4(acc[0], acc[1], acc[2], acc[3]);
      *(float4*)(op + 4) = make_float4(acc[4], acc[5], acc[6], acc[7]);
    }
  }
}

// ---------------------------------------------------------------------------
// Fused LN + scoring MLP: one wave per token (128 cols -> 4/lane).
// s = silu(LN(t) @ w1.T + b1) @ w2.T + b2 ;  hidden dim 32 == wave32 lanes.
// ---------------------------------------------------------------------------
__global__ __launch_bounds__(256) void score_kernel(
    const float* __restrict__ t, const float* __restrict__ g, const float* __restrict__ bb,
    const float* __restrict__ w1, const float* __restrict__ b1,
    const float* __restrict__ w2, const float* __restrict__ b2,
    float* __restrict__ sc, int rows)
{
  const int row  = (blockIdx.x * blockDim.x + threadIdx.x) >> 5;
  const int lane = threadIdx.x & 31;
  if (row >= rows) return;
  const float4 v = ((const float4*)(t + (size_t)row * DMODEL))[lane];
  float s = v.x + v.y + v.z + v.w;
  #pragma unroll
  for (int o = 16; o > 0; o >>= 1) s += __shfl_xor(s, o, 32);
  const float m = s * (1.0f / 128.0f);
  const float d0 = v.x - m, d1 = v.y - m, d2 = v.z - m, d3 = v.w - m;
  float q = d0 * d0 + d1 * d1 + d2 * d2 + d3 * d3;
  #pragma unroll
  for (int o = 16; o > 0; o >>= 1) q += __shfl_xor(q, o, 32);
  const float inv = rsqrtf(q * (1.0f / 128.0f) + 1e-5f);
  const int db = lane * 4;
  const float l0 = d0 * inv * g[db]     + bb[db];
  const float l1 = d1 * inv * g[db + 1] + bb[db + 1];
  const float l2 = d2 * inv * g[db + 2] + bb[db + 2];
  const float l3 = d3 * inv * g[db + 3] + bb[db + 3];
  float acc = b1[lane];
  const float* wrow = w1 + lane * DMODEL;
  for (int src = 0; src < 32; ++src) {
    const float e0 = __shfl(l0, src, 32);
    const float e1 = __shfl(l1, src, 32);
    const float e2 = __shfl(l2, src, 32);
    const float e3 = __shfl(l3, src, 32);
    acc += e0 * wrow[src * 4] + e1 * wrow[src * 4 + 1] +
           e2 * wrow[src * 4 + 2] + e3 * wrow[src * 4 + 3];
  }
  const float h = acc / (1.0f + __expf(-acc));   // silu
  float p = h * w2[lane];
  #pragma unroll
  for (int o = 16; o > 0; o >>= 1) p += __shfl_xor(p, o, 32);
  if (lane == 0) sc[row] = p + b2[0];
}

// ---------------------------------------------------------------------------
// Exact top-24 per batch row: scores in LDS, 24 argmax passes,
// tie-break = smaller index (matches jax.lax.top_k).
// ---------------------------------------------------------------------------
__global__ void topk_kernel(const float* __restrict__ sc, int* __restrict__ fidx, int S)
{
  __shared__ float vals[6400];
  __shared__ float rv[256];
  __shared__ int   ri[256];
  const int b = blockIdx.x;
  const int tid = threadIdx.x;
  for (int i = tid; i < S; i += blockDim.x) vals[i] = sc[(size_t)b * S + i];
  __syncthreads();
  for (int it = 0; it < KF; ++it) {
    float bv = -1e38f; int bi = S;
    for (int i = tid; i < S; i += blockDim.x) {
      const float xv = vals[i];
      if (xv > bv || (xv == bv && i < bi)) { bv = xv; bi = i; }
    }
    rv[tid] = bv; ri[tid] = bi;
    __syncthreads();
    if (tid == 0) {
      float gv = rv[0]; int gi = ri[0];
      for (int i2 = 1; i2 < 256; ++i2)
        if (rv[i2] > gv || (rv[i2] == gv && ri[i2] < gi)) { gv = rv[i2]; gi = ri[i2]; }
      fidx[b * KF + it] = gi;
      vals[gi] = -1e38f;
    }
    __syncthreads();
  }
}

// ---------------------------------------------------------------------------
// Gather / scatter of selected token rows.
// ---------------------------------------------------------------------------
__global__ void gather_coarse_kernel(const float* __restrict__ t0, const float* __restrict__ t1,
                                     const float* __restrict__ t2, const int* __restrict__ fidx,
                                     float* __restrict__ coarse)
{
  const int gb = blockIdx.x;               // B*24
  const int b = gb / NTOK, j = gb % NTOK;
  const int scale = j >> 3, jj = j & 7;    // top-8 per scale, concat order 0,1,2
  const int idx = fidx[(scale * BATCH + b) * KF + jj];
  const float* tp; int S;
  if (scale == 0)      { tp = t0; S = 6400; }
  else if (scale == 1) { tp = t1; S = 1600; }
  else                 { tp = t2; S = 400;  }
  coarse[(size_t)gb * DMODEL + threadIdx.x] =
      tp[((size_t)b * S + idx) * DMODEL + threadIdx.x];
}

__global__ void gather_rows_kernel(const float* __restrict__ t, const int* __restrict__ fidx,
                                   float* __restrict__ dst, int S)
{
  const int gb = blockIdx.x; const int b = gb / KF, j = gb % KF;
  const int idx = fidx[b * KF + j];
  dst[(size_t)gb * DMODEL + threadIdx.x] =
      t[((size_t)b * S + idx) * DMODEL + threadIdx.x];
}

__global__ void scatter_add_kernel(float* __restrict__ t, const int* __restrict__ fidx,
                                   const float* __restrict__ u, int S)
{
  const int gb = blockIdx.x; const int b = gb / KF, j = gb % KF;
  const int idx = fidx[b * KF + j];           // unique per (b) -> no atomics needed
  t[((size_t)b * S + idx) * DMODEL + threadIdx.x] +=
      u[(size_t)gb * DMODEL + threadIdx.x];
}

// ---------------------------------------------------------------------------
// Generic small linear: Y = act(X @ W^T + bias) (+ res). W is [N,K] row-major.
// ---------------------------------------------------------------------------
__global__ void linear_kernel(const float* __restrict__ X, const float* __restrict__ W,
                              const float* __restrict__ bias, const float* __restrict__ res,
                              float* __restrict__ Y, int R, int K, int N, int silu)
{
  const int id = blockIdx.x * blockDim.x + threadIdx.x;
  if (id >= R * N) return;
  const int rrow = id / N, n = id % N;
  const float* xr = X + (size_t)rrow * K;
  const float* wr = W + (size_t)n * K;
  float acc = bias[n];
  for (int k = 0; k < K; ++k) acc += xr[k] * wr[k];
  if (silu) acc = acc / (1.0f + __expf(-acc));
  if (res) acc += res[id];
  Y[id] = acc;
}

__global__ void ln_kernel(const float* __restrict__ X, const float* __restrict__ g,
                          const float* __restrict__ bb, float* __restrict__ Y, int R)
{
  const int row  = (blockIdx.x * blockDim.x + threadIdx.x) >> 5;
  const int lane = threadIdx.x & 31;
  if (row >= R) return;
  const float4 v = ((const float4*)(X + (size_t)row * DMODEL))[lane];
  float s = v.x + v.y + v.z + v.w;
  #pragma unroll
  for (int o = 16; o > 0; o >>= 1) s += __shfl_xor(s, o, 32);
  const float m = s * (1.0f / 128.0f);
  const float d0 = v.x - m, d1 = v.y - m, d2 = v.z - m, d3 = v.w - m;
  float q = d0 * d0 + d1 * d1 + d2 * d2 + d3 * d3;
  #pragma unroll
  for (int o = 16; o > 0; o >>= 1) q += __shfl_xor(q, o, 32);
  const float inv = rsqrtf(q * (1.0f / 128.0f) + 1e-5f);
  const int db = lane * 4;
  float4 ov;
  ov.x = d0 * inv * g[db]     + bb[db];
  ov.y = d1 * inv * g[db + 1] + bb[db + 1];
  ov.z = d2 * inv * g[db + 2] + bb[db + 2];
  ov.w = d3 * inv * g[db + 3] + bb[db + 3];
  ((float4*)(Y + (size_t)row * DMODEL))[lane] = ov;
}

// ---------------------------------------------------------------------------
// MHA core on 24-row sequences: one wave per (batch, head), head_dim = 32.
// Q,K,V,O all [B,24,128]; head h occupies cols [h*32, h*32+32).
// ---------------------------------------------------------------------------
__global__ void attn_kernel(const float* __restrict__ Q, const float* __restrict__ Km,
                            const float* __restrict__ Vm, float* __restrict__ O,
                            int NQ, int NK)
{
  const int bh = blockIdx.x;
  const int b = bh >> 2, h = bh & 3;
  const int lane = threadIdx.x;
  __shared__ float ks[NTOK * 32], vs[NTOK * 32], qs[32], as[32];
  for (int i = lane; i < NK * 32; i += 32) {
    const int rowk = i >> 5, d = i & 31;
    const size_t off = ((size_t)b * NTOK + rowk) * DMODEL + h * 32 + d;
    ks[i] = Km[off]; vs[i] = Vm[off];
  }
  __syncthreads();
  const float scl = 0.17677669529663689f;   // 32^-0.5
  for (int i = 0; i < NQ; ++i) {
    qs[lane] = Q[((size_t)b * NTOK + i) * DMODEL + h * 32 + lane];
    __syncthreads();
    float sj = -1e38f;
    if (lane < NK) {
      float acc = 0.0f;
      for (int d = 0; d < 32; ++d) acc += qs[d] * ks[lane * 32 + d];
      sj = acc * scl;
    }
    float mx = sj;
    #pragma unroll
    for (int o = 16; o > 0; o >>= 1) mx = fmaxf(mx, __shfl_xor(mx, o, 32));
    float p = (lane < NK) ? __expf(sj - mx) : 0.0f;
    float ssum = p;
    #pragma unroll
    for (int o = 16; o > 0; o >>= 1) ssum += __shfl_xor(ssum, o, 32);
    as[lane] = p / ssum;
    __syncthreads();
    float od = 0.0f;
    for (int j = 0; j < NK; ++j) od += as[j] * vs[j * 32 + lane];
    O[((size_t)b * NTOK + i) * DMODEL + h * 32 + lane] = od;
    __syncthreads();
  }
}

// ---------------------------------------------------------------------------
extern "C" void kernel_launch(void* const* d_in, const int* in_sizes, int n_in,
                              void* d_out, int out_size, void* d_ws, size_t ws_size,
                              hipStream_t stream)
{
  const float* p3    = (const float*)d_in[0];
  const float* p4    = (const float*)d_in[1];
  const float* p5    = (const float*)d_in[2];
  const float* win0  = (const float*)d_in[3];
  const float* wout0 = (const float*)d_in[4];
  const float* win1  = (const float*)d_in[5];
  const float* wout1 = (const float*)d_in[6];
  const float* win2  = (const float*)d_in[7];
  const float* wout2 = (const float*)d_in[8];
  const float* ts_w1 = (const float*)d_in[9];
  const float* ts_b1 = (const float*)d_in[10];
  const float* ts_w2 = (const float*)d_in[11];
  const float* ts_b2 = (const float*)d_in[12];
  const float* ng    = (const float*)d_in[13];
  const float* nb    = (const float*)d_in[14];
  const float* ca_wq = (const float*)d_in[15];
  const float* ca_bq = (const float*)d_in[16];
  const float* ca_wk = (const float*)d_in[17];
  const float* ca_bk = (const float*)d_in[18];
  const float* ca_wv = (const float*)d_in[19];
  const float* ca_bv = (const float*)d_in[20];
  const float* ca_wp = (const float*)d_in[21];
  const float* ca_bp = (const float*)d_in[22];
  const float* xa_wq = (const float*)d_in[23];
  const float* xa_bq = (const float*)d_in[24];
  const float* xa_wk = (const float*)d_in[25];
  const float* xa_bk = (const float*)d_in[26];
  const float* xa_wv = (const float*)d_in[27];
  const float* xa_bv = (const float*)d_in[28];
  const float* xa_wp = (const float*)d_in[29];
  const float* xa_bp = (const float*)d_in[30];
  const float* ffn_w1 = (const float*)d_in[31];
  const float* ffn_b1 = (const float*)d_in[32];
  const float* ffn_w2 = (const float*)d_in[33];
  const float* ffn_b2 = (const float*)d_in[34];
  (void)in_sizes; (void)n_in; (void)out_size; (void)ws_size;

  constexpr int S0 = 6400, S1 = 1600, S2 = 400;
  constexpr int C0 = 256,  C1 = 512,  C2 = 1024;

  unsigned char* wp = (unsigned char*)d_ws;
  auto carve = [&](size_t bytes) -> float* {
    float* p = (float*)wp;
    wp += (bytes + 255) & ~(size_t)255;
    return p;
  };
  float* t0  = carve((size_t)BATCH * S0 * DMODEL * 4);
  float* t1  = carve((size_t)BATCH * S1 * DMODEL * 4);
  float* t2  = carve((size_t)BATCH * S2 * DMODEL * 4);
  float* sc0 = carve((size_t)BATCH * S0 * 4);
  float* sc1 = carve((size_t)BATCH * S1 * 4);
  float* sc2 = carve((size_t)BATCH * S2 * 4);
  int*   fidx = (int*)carve((size_t)3 * BATCH * KF * 4);
  const size_t TOKB = (size_t)BATCH * NTOK * DMODEL * 4;
  float* coarse = carve(TOKB);
  float* qb  = carve(TOKB); float* kb  = carve(TOKB); float* vb = carve(TOKB);
  float* ao  = carve(TOKB); float* cu  = carve(TOKB); float* lnb = carve(TOKB);
  float* cu2 = carve(TOKB);
  float* f1  = carve((size_t)BATCH * NTOK * 512 * 4);
  float* kx  = carve(TOKB); float* vx  = carve(TOKB);
  float* fs  = carve(TOKB); float* fq  = carve(TOKB); float* fao = carve(TOKB);
  float* ub  = carve(TOKB); float* u2  = carve(TOKB);

  // Opt in to >64KB dynamic LDS (gfx1250 WGP has 320KB).
  hipFuncSetAttribute((const void*)proj_in_wmma<C0, S0>,
                      hipFuncAttributeMaxDynamicSharedMemorySize, DMODEL * C0 * 2);
  hipFuncSetAttribute((const void*)proj_in_wmma<C1, S1>,
                      hipFuncAttributeMaxDynamicSharedMemorySize, DMODEL * C1 * 2);
  hipFuncSetAttribute((const void*)proj_in_wmma<C2, S2>,
                      hipFuncAttributeMaxDynamicSharedMemorySize, DMODEL * C2 * 2);
  hipFuncSetAttribute((const void*)proj_out_wmma<C0, S0>,
                      hipFuncAttributeMaxDynamicSharedMemorySize, DMODEL * C0 * 2);
  hipFuncSetAttribute((const void*)proj_out_wmma<C1, S1>,
                      hipFuncAttributeMaxDynamicSharedMemorySize, DMODEL * C1 * 2);
  hipFuncSetAttribute((const void*)proj_out_wmma<C2, S2>,
                      hipFuncAttributeMaxDynamicSharedMemorySize, DMODEL * C2 * 2);

  // ---- proj_in (WMMA) ----
  proj_in_wmma<C0, S0><<<(BATCH * S0 / 16 + 7) / 8, 256, DMODEL * C0 * 2, stream>>>(p3, win0, t0);
  proj_in_wmma<C1, S1><<<(BATCH * S1 / 16 + 7) / 8, 256, DMODEL * C1 * 2, stream>>>(p4, win1, t1);
  proj_in_wmma<C2, S2><<<(BATCH * S2 / 16 + 7) / 8, 256, DMODEL * C2 * 2, stream>>>(p5, win2, t2);

  // ---- token scores ----
  score_kernel<<<BATCH * S0 / 8, 256, 0, stream>>>(t0, ng, nb, ts_w1, ts_b1, ts_w2, ts_b2, sc0, BATCH * S0);
  score_kernel<<<BATCH * S1 / 8, 256, 0, stream>>>(t1, ng, nb, ts_w1, ts_b1, ts_w2, ts_b2, sc1, BATCH * S1);
  score_kernel<<<BATCH * S2 / 8, 256, 0, stream>>>(t2, ng, nb, ts_w1, ts_b1, ts_w2, ts_b2, sc2, BATCH * S2);

  // ---- top-24 per (scale, batch) ----
  topk_kernel<<<BATCH, 256, 0, stream>>>(sc0, fidx + 0 * BATCH * KF, S0);
  topk_kernel<<<BATCH, 256, 0, stream>>>(sc1, fidx + 1 * BATCH * KF, S1);
  topk_kernel<<<BATCH, 256, 0, stream>>>(sc2, fidx + 2 * BATCH * KF, S2);

  // ---- coarse self-attention block ----
  const int R = BATCH * NTOK;
  auto lin = [&](const float* X, const float* W, const float* bias, const float* res,
                 float* Y, int K, int N, int silu) {
    const int tot = R * N;
    linear_kernel<<<(tot + 255) / 256, 256, 0, stream>>>(X, W, bias, res, Y, R, K, N, silu);
  };
  gather_coarse_kernel<<<R, DMODEL, 0, stream>>>(t0, t1, t2, fidx, coarse);
  lin(coarse, ca_wq, ca_bq, nullptr, qb, 128, 128, 0);
  lin(coarse, ca_wk, ca_bk, nullptr, kb, 128, 128, 0);
  lin(coarse, ca_wv, ca_bv, nullptr, vb, 128, 128, 0);
  attn_kernel<<<BATCH * 4, 32, 0, stream>>>(qb, kb, vb, ao, NTOK, NTOK);
  lin(ao, ca_wp, ca_bp, nullptr, cu, 128, 128, 0);          // cu = mha(...)  (no residual)
  ln_kernel<<<R * 32 / 256, 256, 0, stream>>>(cu, ng, nb, lnb, R);
  lin(lnb, ffn_w1, ffn_b1, nullptr, f1, 128, 512, 1);
  lin(f1, ffn_w2, ffn_b2, cu, cu2, 512, 128, 0);            // cu2 = ffn(ln(cu)) + cu

  // cross-attention K/V from cu2 (shared by all three scales)
  lin(cu2, xa_wk, xa_bk, nullptr, kx, 128, 128, 0);
  lin(cu2, xa_wv, xa_bv, nullptr, vx, 128, 128, 0);

  // ---- per-scale fine path + proj_out ----
  float* outs0 = (float*)d_out;
  float* outs1 = outs0 + (size_t)BATCH * C0 * S0;
  float* outs2 = outs1 + (size_t)BATCH * C1 * S1;

  // scale 0
  {
    const int* fi = fidx + 0 * BATCH * KF;
    gather_rows_kernel<<<R, DMODEL, 0, stream>>>(t0, fi, fs, S0);
    lin(fs, xa_wq, xa_bq, nullptr, fq, 128, 128, 0);
    attn_kernel<<<BATCH * 4, 32, 0, stream>>>(fq, kx, vx, fao, NTOK, NTOK);
    lin(fao, xa_wp, xa_bp, nullptr, ub, 128, 128, 0);       // u = mha(fs, cu, cu)
    ln_kernel<<<R * 32 / 256, 256, 0, stream>>>(ub, ng, nb, lnb, R);
    lin(lnb, ffn_w1, ffn_b1, nullptr, f1, 128, 512, 1);
    lin(f1, ffn_w2, ffn_b2, ub, u2, 512, 128, 0);           // u2 = ffn(ln(u)) + u
    scatter_add_kernel<<<R, DMODEL, 0, stream>>>(t0, fi, u2, S0);
    proj_out_wmma<C0, S0><<<(BATCH * S0 / 16 + 7) / 8, 256, DMODEL * C0 * 2, stream>>>(t0, wout0, outs0);
  }
  // scale 1
  {
    const int* fi = fidx + 1 * BATCH * KF;
    gather_rows_kernel<<<R, DMODEL, 0, stream>>>(t1, fi, fs, S1);
    lin(fs, xa_wq, xa_bq, nullptr, fq, 128, 128, 0);
    attn_kernel<<<BATCH * 4, 32, 0, stream>>>(fq, kx, vx, fao, NTOK, NTOK);
    lin(fao, xa_wp, xa_bp, nullptr, ub, 128, 128, 0);
    ln_kernel<<<R * 32 / 256, 256, 0, stream>>>(ub, ng, nb, lnb, R);
    lin(lnb, ffn_w1, ffn_b1, nullptr, f1, 128, 512, 1);
    lin(f1, ffn_w2, ffn_b2, ub, u2, 512, 128, 0);
    scatter_add_kernel<<<R, DMODEL, 0, stream>>>(t1, fi, u2, S1);
    proj_out_wmma<C1, S1><<<(BATCH * S1 / 16 + 7) / 8, 256, DMODEL * C1 * 2, stream>>>(t1, wout1, outs1);
  }
  // scale 2
  {
    const int* fi = fidx + 2 * BATCH * KF;
    gather_rows_kernel<<<R, DMODEL, 0, stream>>>(t2, fi, fs, S2);
    lin(fs, xa_wq, xa_bq, nullptr, fq, 128, 128, 0);
    attn_kernel<<<BATCH * 4, 32, 0, stream>>>(fq, kx, vx, fao, NTOK, NTOK);
    lin(fao, xa_wp, xa_bp, nullptr, ub, 128, 128, 0);
    ln_kernel<<<R * 32 / 256, 256, 0, stream>>>(ub, ng, nb, lnb, R);
    lin(lnb, ffn_w1, ffn_b1, nullptr, f1, 128, 512, 1);
    lin(f1, ffn_w2, ffn_b2, ub, u2, 512, 128, 0);
    scatter_add_kernel<<<R, DMODEL, 0, stream>>>(t2, fi, u2, S2);
    proj_out_wmma<C2, S2><<<(BATCH * S2 / 16 + 7) / 8, 256, DMODEL * C2 * 2, stream>>>(t2, wout2, outs2);
  }
}
